// MyRNN_73839077753095
// MI455X (gfx1250) — matrix operand order repaired and checked by reading, not compiled, not run on hardware
//
#include <hip/hip_runtime.h>
#include <hip/hip_bf16.h>

#define B_ 32
#define L_ 2048
#define D_ 256
#define H_ 512
#define NB_REC 4   // persistent blocks in the recurrence grid (must be co-resident)

typedef __bf16 v16bf __attribute__((ext_vector_type(16)));
typedef float  v8f   __attribute__((ext_vector_type(8)));

union BF16x16 { v16bf v; __bf16 e[16]; };

__device__ __forceinline__ v8f wmma_bf16(v16bf a, v16bf b, v8f c) {
  // D = A(16x32 bf16) * B(32x16 bf16) + C(16x16 f32)
  return __builtin_amdgcn_wmma_f32_16x16x32_bf16(false, a, false, b, (short)0, c,
                                                 false, false);
}

// ---------------------------------------------------------------------------
// Pack W (N=512 rows x K cols, row-major; used as B = W^T, i.e. B[k][n]=W[n][k])
// into the CDNA5 WMMA B-operand layout, one 32x16 K-x-N tile = 512 bf16:
//   lane l: column n = nt*16 + (l&15);  element j: row k = kt*32 + 8*(l>>4) + (j&7) + 16*(j>>3)
// Block index = kt*(N/16) + nt; element index = block*512 + lane*16 + j.
// ---------------------------------------------------------------------------
__device__ __forceinline__ void pack_weight(__bf16* dst, const float* W, int K, int e) {
  int block  = e >> 9;
  int within = e & 511;
  int lane = within >> 4;
  int j    = within & 15;
  int nt = block & 31;          // N/16 == 32 for all weights here
  int kt = block >> 5;
  int n = nt * 16 + (lane & 15);
  int k = kt * 32 + 8 * (lane >> 4) + (j & 7) + 16 * (j >> 3);
  dst[e] = (__bf16)W[(size_t)n * K + k];
}

__global__ void prep_kernel(const float* __restrict__ Whx,
                            const float* __restrict__ Whh,
                            const float* __restrict__ Woh,
                            __bf16* __restrict__ wxp,
                            __bf16* __restrict__ whhp,
                            __bf16* __restrict__ wohp,
                            __bf16* __restrict__ hpack,
                            unsigned* __restrict__ bar) {
  const int WX = (D_ / 32) * (H_ / 16) * 512;       // 131072
  const int WH = (H_ / 32) * (H_ / 16) * 512;       // 262144
  const int HP = 2 * ((B_ / 16) * (H_ / 32) * 512); // 32768 (ping-pong h pack)
  int e = blockIdx.x * 256 + threadIdx.x;
  if (e < WX) { pack_weight(wxp, Whx, D_, e); return; }
  e -= WX;
  if (e < WH) { pack_weight(whhp, Whh, H_, e); return; }
  e -= WH;
  if (e < WH) { pack_weight(wohp, Woh, H_, e); return; }
  e -= WH;
  if (e < HP) { hpack[e] = (__bf16)0.0f; return; }   // h0 = 0
  e -= HP;
  if (e < 2) bar[e] = 0u;                            // barrier count / generation
}

// ---------------------------------------------------------------------------
// C = A(f32 row-major, M x K) * Bpack + bias. One wave per 16x16 output tile.
// A gathered per the 16-bit WMMA A layout: lane row m = m0+(l&15),
// K elements j<8 -> base+j, j>=8 -> base+16+(j-8), base = kt*32 + 8*(l>>4):
// two contiguous 8-float runs -> 4x float4 loads, cvt to bf16 in-register.
// Output columns are H_-wide. K-loop unrolled 2x; branchless clamped prefetch.
// ---------------------------------------------------------------------------
__global__ void gemm_bias_kernel(const float* __restrict__ A, int lda, int ksteps,
                                 const __bf16* __restrict__ Bp,
                                 const float* __restrict__ bias,
                                 float* __restrict__ C, int mtiles) {
  int wid  = (blockIdx.x * blockDim.x + threadIdx.x) >> 5;
  int lane = threadIdx.x & 31;
  if (wid >= mtiles * 32) return;
  int nt    = wid & 31;
  int mtile = wid >> 5;
  int m0 = mtile << 4;
  int hi = lane >> 4, lo = lane & 15;
  int n  = nt * 16 + lo;
  const float* arow = A + (size_t)(m0 + lo) * lda;
  v8f c = {};
#pragma unroll 2
  for (int kt = 0; kt < ksteps; ++kt) {
    int base = kt * 32 + 8 * hi;
    float4 f0 = *(const float4*)(arow + base);
    float4 f1 = *(const float4*)(arow + base + 4);
    float4 g0 = *(const float4*)(arow + base + 16);
    float4 g1 = *(const float4*)(arow + base + 20);
    // branchless clamped prefetch of the next B tile (no in-loop diamond)
    int pkt = kt + 1 < ksteps ? kt + 1 : kt;   // s_min-style select, wave-uniform
    __builtin_prefetch(Bp + ((size_t)(pkt * 32 + nt) << 9) + lane * 16, 0, 3);
    BF16x16 a;
    a.e[0]  = (__bf16)f0.x; a.e[1]  = (__bf16)f0.y; a.e[2]  = (__bf16)f0.z; a.e[3]  = (__bf16)f0.w;
    a.e[4]  = (__bf16)f1.x; a.e[5]  = (__bf16)f1.y; a.e[6]  = (__bf16)f1.z; a.e[7]  = (__bf16)f1.w;
    a.e[8]  = (__bf16)g0.x; a.e[9]  = (__bf16)g0.y; a.e[10] = (__bf16)g0.z; a.e[11] = (__bf16)g0.w;
    a.e[12] = (__bf16)g1.x; a.e[13] = (__bf16)g1.y; a.e[14] = (__bf16)g1.z; a.e[15] = (__bf16)g1.w;
    v16bf b = *(const v16bf*)(Bp + ((size_t)(kt * 32 + nt) << 9) + lane * 16);
    c = wmma_bf16(a.v, b, c);
  }
  float bn = bias[n];
#pragma unroll
  for (int r = 0; r < 8; ++r) {
    int gm = m0 + r + 8 * hi;                 // C layout: VGPR r <-> row r + 8*hi
    C[(size_t)gm * H_ + n] = c[r] + bn;
  }
}

// ---------------------------------------------------------------------------
// Persistent recurrence: for t in 0..L-1:
//   h_t = relu(xp_t + h_{t-1} @ Whh^T + bhh)     (xp already in hid[])
// 64 waves (4 blocks x 16 waves); wave (mt,nt) owns the 16x16 tile of the
// 32x512 h matrix. h carried step-to-step as bf16 in the WMMA A-pack layout
// (ping-pong buffers), device-scope software barrier per step.
// ---------------------------------------------------------------------------
__global__ void __launch_bounds__(512, 1)
recurrence_kernel(const __bf16* __restrict__ whhp,
                  const float* __restrict__ bhh,
                  float* __restrict__ hid,        // (B*L, H): xp in, h out
                  __bf16* __restrict__ hpack,     // 2 x 16384 bf16
                  unsigned* __restrict__ bar) {
  const int HPE = (B_ / 16) * (H_ / 32) * 512;    // 16384 per buffer
  int w    = (blockIdx.x << 4) + (threadIdx.x >> 5);
  int lane = threadIdx.x & 31;
  int nt = w & 31, mt = w >> 5;
  int hi = lane >> 4, lo = lane & 15;
  int n  = nt * 16 + lo;
  float bn = bhh[n];
  // Scatter params: value h[m][k=n] -> A-pack block (mt, kt_p), lane lp, elem j
  int kt_p    = nt >> 1;
  int k_local = lo + 16 * (nt & 1);
  int hi_p = (k_local >> 3) & 1;
  int jj   = k_local - 8 * hi_p - 8 * (k_local >> 4);
  size_t rowbase[8];
#pragma unroll
  for (int r = 0; r < 8; ++r)
    rowbase[r] = (size_t)(mt * 16 + r + 8 * hi) * L_;

  for (int t = 0; t < L_; ++t) {
    const __bf16* apack = hpack + (size_t)(t & 1) * HPE;
    __bf16* npack       = hpack + (size_t)((t + 1) & 1) * HPE;
    v8f c = {};
#pragma unroll 4
    for (int kt = 0; kt < 16; ++kt) {
      v16bf a = *(const v16bf*)(apack + ((((mt << 4) + kt) << 9) + lane * 16));
      v16bf b = *(const v16bf*)(whhp + (((size_t)((kt << 5) + nt) << 9) + lane * 16));
      c = wmma_bf16(a, b, c);
    }
#pragma unroll
    for (int r = 0; r < 8; ++r) {
      size_t off = (rowbase[r] + t) * H_ + n;
      float v = c[r] + bn + hid[off];          // + xp_t
      v = v > 0.0f ? v : 0.0f;                 // relu
      hid[off] = v;                            // hidden_arr output (f32)
      int lp = (r + 8 * hi) + 16 * hi_p;       // next-step A-pack slot
      npack[(((mt << 4) + kt_p) << 9) + lp * 16 + jj] = (__bf16)v;
    }
    // ---- device-scope barrier (generation = t+1, monotonic counter) ----
    __threadfence();
    __syncthreads();
    if (threadIdx.x == 0) {
      unsigned arr = __hip_atomic_fetch_add(&bar[0], 1u, __ATOMIC_ACQ_REL,
                                            __HIP_MEMORY_SCOPE_AGENT) + 1u;
      if (arr == (unsigned)(t + 1) * NB_REC)
        __hip_atomic_store(&bar[1], (unsigned)(t + 1), __ATOMIC_RELEASE,
                           __HIP_MEMORY_SCOPE_AGENT);
      while (__hip_atomic_load(&bar[1], __ATOMIC_ACQUIRE,
                               __HIP_MEMORY_SCOPE_AGENT) < (unsigned)(t + 1))
        __builtin_amdgcn_s_sleep(2);
    }
    __syncthreads();
    __threadfence();
  }
}

// ---------------------------------------------------------------------------
// In-place softmax over each 512-wide row; one wave per row, 16 vals/lane.
// ---------------------------------------------------------------------------
__global__ void softmax_kernel(float* __restrict__ Y, int rows) {
  int wid  = (blockIdx.x * blockDim.x + threadIdx.x) >> 5;
  int lane = threadIdx.x & 31;
  if (wid >= rows) return;
  float* p = Y + (size_t)wid * H_;
  float v[16];
  float mx = -3.0e38f;
#pragma unroll
  for (int i = 0; i < 16; ++i) { v[i] = p[i * 32 + lane]; mx = fmaxf(mx, v[i]); }
#pragma unroll
  for (int off = 16; off > 0; off >>= 1) mx = fmaxf(mx, __shfl_xor(mx, off, 32));
  float s = 0.0f;
#pragma unroll
  for (int i = 0; i < 16; ++i) { v[i] = __expf(v[i] - mx); s += v[i]; }
#pragma unroll
  for (int off = 16; off > 0; off >>= 1) s += __shfl_xor(s, off, 32);
  float inv = 1.0f / s;
#pragma unroll
  for (int i = 0; i < 16; ++i) p[i * 32 + lane] = v[i] * inv;
}

extern "C" void kernel_launch(void* const* d_in, const int* in_sizes, int n_in,
                              void* d_out, int out_size, void* d_ws, size_t ws_size,
                              hipStream_t stream) {
  (void)in_sizes; (void)n_in; (void)out_size; (void)ws_size;
  const float* x   = (const float*)d_in[0];
  const float* Whx = (const float*)d_in[1];
  const float* bhx = (const float*)d_in[2];
  const float* Whh = (const float*)d_in[3];
  const float* bhh = (const float*)d_in[4];
  const float* Woh = (const float*)d_in[5];
  const float* boh = (const float*)d_in[6];

  float* y   = (float*)d_out;                       // output_arr (B,L,H)
  float* hid = y + (size_t)B_ * L_ * H_;            // hidden_arr (B,L,H)

  char* ws = (char*)d_ws;                           // ~1.4 MB used
  unsigned* bar = (unsigned*)ws;                                    // 256 B
  __bf16* wxp  = (__bf16*)(ws + 256);                               // 256 KB
  __bf16* whhp = (__bf16*)(ws + 256 + 262144);                      // 512 KB
  __bf16* wohp = (__bf16*)(ws + 256 + 262144 + 524288);             // 512 KB
  __bf16* hpk  = (__bf16*)(ws + 256 + 262144 + 524288 + 524288);    // 64 KB

  const int MT = (B_ * L_) / 16;                    // 4096 M-tiles

  // 1) pack weights to bf16 WMMA-B layout, zero h0 pack + barrier state
  int prep_total = 131072 + 262144 + 262144 + 32768 + 2;
  prep_kernel<<<(prep_total + 255) / 256, 256, 0, stream>>>(
      Whx, Whh, Woh, wxp, whhp, wohp, hpk, bar);

  // 2) xp = x @ Whx^T + bhx  ->  hid region (consumed+overwritten by scan)
  gemm_bias_kernel<<<MT * 4, 256, 0, stream>>>(x, D_, D_ / 32, wxp, bhx, hid, MT);

  // 3) sequential scan: h_t = relu(xp_t + h_{t-1} @ Whh^T + bhh), in place
  recurrence_kernel<<<NB_REC, 512, 0, stream>>>(whhp, bhh, hid, hpk, bar);

  // 4) logits = h @ Woh^T + boh -> y region
  gemm_bias_kernel<<<MT * 4, 256, 0, stream>>>(hid, H_, H_ / 32, wohp, boh, y, MT);

  // 5) softmax in place over H
  softmax_kernel<<<(B_ * L_) / 8, 256, 0, stream>>>(y, B_ * L_);
}
